// OuroborosMoE_36833639530922
// MI455X (gfx1250) — matrix operand order repaired and checked by reading, not compiled, not run on hardware
//
#include <hip/hip_runtime.h>
#include <hip/hip_bf16.h>
#include <math.h>

// Problem constants (B=4, T=1024, D=1024, E=8, K=2, H=4096)
#define NTOK 4096           // B*T
#define DDIM 1024
#define HDIM 4096
#define NEXP 8
#define MTILE 32            // tokens per block
#define TILES (NTOK / MTILE)   // 128 worst-case tile rows per expert
#define HCHUNK 256          // H processed per outer iteration

#if __has_builtin(__builtin_amdgcn_global_load_async_to_lds_b128) && \
    __has_builtin(__builtin_amdgcn_s_wait_asynccnt)
#define USE_ASYNC_LDS 1
#else
#define USE_ASYNC_LDS 0
#endif

typedef __attribute__((ext_vector_type(16))) __bf16 v16bf;
typedef __attribute__((ext_vector_type(8)))  float  v8f;
typedef __attribute__((ext_vector_type(4)))  int    v4i;

// Exact pointer types for the async-LDS builtin (from hipcc diagnostic:
// param0 is "int __vector(4) __device__ *", i.e. v4i in AS1; LDS side AS3).
#define ASYNC_GPTR(p) ((__attribute__((address_space(1))) v4i*) \
                       (__attribute__((address_space(1))) void*)(void*)(p))
#define ASYNC_SPTR(p) ((__attribute__((address_space(3))) v4i*) \
                       (__attribute__((address_space(3))) void*)(void*)(p))

union ABpack {
    v16bf v;
    unsigned short s[16];
    uint4 q[2];
};

__device__ __forceinline__ unsigned short f32_to_bf16(float f) {
    union { float f; unsigned int u; } x; x.f = f;
    unsigned int u = x.u;
    u += 0x7FFFu + ((u >> 16) & 1u);   // round-to-nearest-even
    return (unsigned short)(u >> 16);
}

// ---------------------------------------------------------------------------
// Router: softmax over top-2, merge duplicate expert picks.
// ---------------------------------------------------------------------------
__global__ void router_kernel(const float* __restrict__ ew,
                              const int* __restrict__ eidx,
                              int* __restrict__ e0, int* __restrict__ e1,
                              float* __restrict__ c0, float* __restrict__ c1) {
    int n = blockIdx.x * blockDim.x + threadIdx.x;
    if (n >= NTOK) return;
    float a0 = ew[2 * n], a1 = ew[2 * n + 1];
    float m  = fmaxf(a0, a1);
    float w0 = expf(a0 - m), w1 = expf(a1 - m);
    float inv = 1.0f / (w0 + w1);
    w0 *= inv; w1 *= inv;
    int i0 = eidx[2 * n], i1 = eidx[2 * n + 1];
    if (i0 == i1) { e0[n] = i0; c0[n] = 1.0f; e1[n] = -1; c1[n] = 0.0f; }
    else          { e0[n] = i0; c0[n] = w0;   e1[n] = i1; c1[n] = w1;   }
}

// ---------------------------------------------------------------------------
// Deterministic per-(slot,expert) token lists (token order preserved).
// ---------------------------------------------------------------------------
__global__ void build_lists_kernel(const int* __restrict__ e0,
                                   const int* __restrict__ e1,
                                   int* __restrict__ lists,
                                   int* __restrict__ counts) {
    int t = threadIdx.x;                 // 0..15 -> (slot, expert)
    if (t >= 2 * NEXP) return;
    int slot = t >> 3, e = t & 7;
    const int* src = slot ? e1 : e0;
    int cnt = 0;
    int base = t * NTOK;
    for (int n = 0; n < NTOK; ++n)
        if (src[n] == e) lists[base + cnt++] = n;
    counts[t] = cnt;
}

// ---------------------------------------------------------------------------
// x -> bf16 staging buffer.
// ---------------------------------------------------------------------------
__global__ void cast_x_kernel(const float* __restrict__ x,
                              unsigned short* __restrict__ xb) {
    int stride = gridDim.x * blockDim.x;
    for (int i = blockIdx.x * blockDim.x + threadIdx.x; i < NTOK * DDIM; i += stride)
        xb[i] = f32_to_bf16(x[i]);
}

// ---------------------------------------------------------------------------
// Weight prep: f32 [E][R][C] -> bf16 transposed [E][C][R].
// Puts the GEMM reduction dim contiguous so B tiles load as 2x b128/lane.
// ---------------------------------------------------------------------------
__global__ __launch_bounds__(256) void transpose_cast_kernel(
    const float* __restrict__ src, unsigned short* __restrict__ dst,
    int R, int C) {
    __shared__ unsigned short tile[32][33];     // pad -> no bank conflicts
    const size_t eoff = (size_t)blockIdx.z * R * C;
    const float* s = src + eoff;
    unsigned short* d = dst + eoff;
    const int c0 = blockIdx.x * 32;
    const int r0 = blockIdx.y * 32;
    const int tx = threadIdx.x & 31;
    const int ty = threadIdx.x >> 5;            // 0..7
    #pragma unroll
    for (int i = 0; i < 4; ++i) {
        int r = ty + i * 8;
        tile[r][tx] = f32_to_bf16(s[(size_t)(r0 + r) * C + c0 + tx]);
    }
    __syncthreads();
    #pragma unroll
    for (int i = 0; i < 4; ++i) {
        int r = ty + i * 8;                     // output row = source column
        d[(size_t)(c0 + r) * R + r0 + tx] = tile[tx][r];
    }
}

// ---------------------------------------------------------------------------
// B-operand tile (lane&15 = column, 16 halves = contiguous K run from kbase).
// PRECONV: bf16 transposed weights -> two b128 loads.  Fallback: f32 gather.
// ---------------------------------------------------------------------------
template <bool PRECONV>
__device__ __forceinline__ v16bf load_b_tile(const unsigned short* __restrict__ bt,
                                             const float* __restrict__ bf,
                                             int n, int kbase, int ldT, int ldF) {
    ABpack Bp;
    if (PRECONV) {
        const unsigned short* p = bt + (size_t)n * ldT + kbase;
        Bp.q[0] = *((const uint4*)p);
        Bp.q[1] = *((const uint4*)(p + 8));
    } else {
        const float* p = bf + (size_t)kbase * ldF + n;
        #pragma unroll
        for (int j = 0; j < 16; ++j)
            Bp.s[j] = f32_to_bf16(p[(size_t)j * ldF]);
    }
    return Bp.v;
}

// ---------------------------------------------------------------------------
// Fused expert FFN for one routing slot.
//   slot 0: out = x + c0 * FFN_{e0}(x)      (covers every token)
//   slot 1: out += c1 * FFN_{e1}(x)         (tokens with distinct 2nd expert)
// Block: 256 threads = 8 wave32; 32 tokens x 1024 output cols per block.
// ---------------------------------------------------------------------------
template <bool PRECONV>
__global__ __launch_bounds__(256) void moe_ffn_kernel(
    int slot,
    const unsigned short* __restrict__ xb,    // [NTOK][DDIM] bf16
    const float* __restrict__ xin,            // [NTOK][DDIM] f32 (residual)
    const float* __restrict__ W1,             // [E][DDIM][HDIM] f32
    const unsigned short* __restrict__ W1t,   // [E][HDIM][DDIM] bf16 (PRECONV)
    const float* __restrict__ b1,             // [E][HDIM]
    const float* __restrict__ W2,             // [E][HDIM][DDIM] f32
    const unsigned short* __restrict__ W2t,   // [E][DDIM][HDIM] bf16 (PRECONV)
    const float* __restrict__ b2,             // [E][DDIM]
    const int* __restrict__ lists,            // [2*E][NTOK]
    const int* __restrict__ counts,           // [2*E]
    const float* __restrict__ c0,
    const float* __restrict__ c1,
    float* __restrict__ out)
{
    __shared__ int   toks[MTILE];
    __shared__ float cs[MTILE];
    __shared__ unsigned short xs[MTILE][DDIM];     // 64 KB bf16 x tile
    __shared__ unsigned short hs[MTILE][HCHUNK];   // 16 KB bf16 h chunk

    const int e    = blockIdx.x / TILES;
    const int trow = blockIdx.x % TILES;
    const int le   = slot * NEXP + e;
    const int cnt  = counts[le];
    const int m0   = trow * MTILE;
    if (m0 >= cnt) return;                        // uniform per block
    const int rows = min(MTILE, cnt - m0);

    const int tid    = threadIdx.x;
    const int wave   = tid >> 5;                  // 0..7
    const int lane   = tid & 31;
    const int lcol   = lane & 15;
    const bool hiH   = (lane >= 16);
    const int koffB  = hiH ? 16 : 0;              // B-operand K offset (contiguous k order, matches A)
    const int koffA  = hiH ? 16 : 0;              // A-operand K offset (contiguous k order, matches B)
    const int mBase  = hiH ? 8  : 0;              // C-operand row offset (ISA layout)

    if (tid < MTILE) {
        int r  = min(m0 + tid, cnt - 1);          // clamp pad rows (writes masked)
        int tk = lists[le * NTOK + r];
        toks[tid] = tk;
        cs[tid]   = slot ? c1[tk] : c0[tk];
    }
    __syncthreads();

    // Stage 32 token rows of bf16 x into LDS (async DMA path when available).
#if USE_ASYNC_LDS
    for (int i = tid; i < MTILE * (DDIM / 8); i += 256) {
        int r  = i >> 7;                          // DDIM/8 == 128
        int c8 = i & 127;
        __builtin_amdgcn_global_load_async_to_lds_b128(
            ASYNC_GPTR(xb + (size_t)toks[r] * DDIM + c8 * 8),
            ASYNC_SPTR(&xs[r][c8 * 8]),
            0, 0);
    }
    __builtin_amdgcn_s_wait_asynccnt(0);
#else
    for (int i = tid; i < MTILE * (DDIM / 8); i += 256) {
        int r  = i >> 7;
        int c8 = i & 127;
        const uint4* src = (const uint4*)(xb + (size_t)toks[r] * DDIM + c8 * 8);
        *((uint4*)&xs[r][c8 * 8]) = *src;
    }
#endif
    __syncthreads();

    const float*          W1e  = W1  + (size_t)e * DDIM * HDIM;
    const unsigned short* W1te = W1t + (size_t)e * HDIM * DDIM;
    const float*          W2e  = W2  + (size_t)e * HDIM * DDIM;
    const unsigned short* W2te = W2t + (size_t)e * DDIM * HDIM;
    const float* b1e = b1 + (size_t)e * HDIM;
    const float* b2e = b2 + (size_t)e * DDIM;

    v8f acc[2][8];
    #pragma unroll
    for (int r = 0; r < 2; ++r)
        #pragma unroll
        for (int t = 0; t < 8; ++t) acc[r][t] = v8f{};

    for (int hc = 0; hc < HDIM; hc += HCHUNK) {
        // ------- GEMM1: h[0:32, hc + wave*32 .. +32), two col-tiles/wave ----
        v8f hacc[2][2];
        hacc[0][0] = v8f{}; hacc[0][1] = v8f{};
        hacc[1][0] = v8f{}; hacc[1][1] = v8f{};
        const int n1a = hc + wave * 32 + lcol;    // first H column of this lane
        const int n1b = n1a + 16;                 // second col-tile

        // software-pipelined B tiles (double buffer)
        v16bf Ba = load_b_tile<PRECONV>(W1te, W1e, n1a, koffB, DDIM, HDIM);
        v16bf Bb = load_b_tile<PRECONV>(W1te, W1e, n1b, koffB, DDIM, HDIM);
        for (int k0 = 0; k0 < DDIM - 32; k0 += 32) {
            v16bf Na = load_b_tile<PRECONV>(W1te, W1e, n1a, k0 + 32 + koffB, DDIM, HDIM);
            v16bf Nb = load_b_tile<PRECONV>(W1te, W1e, n1b, k0 + 32 + koffB, DDIM, HDIM);
            ABpack A0, A1;
            const int kb = k0 + koffA;
            A0.q[0] = *((const uint4*)&xs[lcol][kb]);
            A0.q[1] = *((const uint4*)&xs[lcol][kb + 8]);
            A1.q[0] = *((const uint4*)&xs[16 + lcol][kb]);
            A1.q[1] = *((const uint4*)&xs[16 + lcol][kb + 8]);
            hacc[0][0] = __builtin_amdgcn_wmma_f32_16x16x32_bf16(
                             false, A0.v, false, Ba, (short)0, hacc[0][0], false, false);
            hacc[1][0] = __builtin_amdgcn_wmma_f32_16x16x32_bf16(
                             false, A1.v, false, Ba, (short)0, hacc[1][0], false, false);
            hacc[0][1] = __builtin_amdgcn_wmma_f32_16x16x32_bf16(
                             false, A0.v, false, Bb, (short)0, hacc[0][1], false, false);
            hacc[1][1] = __builtin_amdgcn_wmma_f32_16x16x32_bf16(
                             false, A1.v, false, Bb, (short)0, hacc[1][1], false, false);
            Ba = Na; Bb = Nb;
        }
        {   // peeled final k-step
            ABpack A0, A1;
            const int kb = (DDIM - 32) + koffA;
            A0.q[0] = *((const uint4*)&xs[lcol][kb]);
            A0.q[1] = *((const uint4*)&xs[lcol][kb + 8]);
            A1.q[0] = *((const uint4*)&xs[16 + lcol][kb]);
            A1.q[1] = *((const uint4*)&xs[16 + lcol][kb + 8]);
            hacc[0][0] = __builtin_amdgcn_wmma_f32_16x16x32_bf16(
                             false, A0.v, false, Ba, (short)0, hacc[0][0], false, false);
            hacc[1][0] = __builtin_amdgcn_wmma_f32_16x16x32_bf16(
                             false, A1.v, false, Ba, (short)0, hacc[1][0], false, false);
            hacc[0][1] = __builtin_amdgcn_wmma_f32_16x16x32_bf16(
                             false, A0.v, false, Bb, (short)0, hacc[0][1], false, false);
            hacc[1][1] = __builtin_amdgcn_wmma_f32_16x16x32_bf16(
                             false, A1.v, false, Bb, (short)0, hacc[1][1], false, false);
        }

        // bias + exact-erf GELU, pack bf16 into LDS h chunk
        const float bba = b1e[n1a];
        const float bbb = b1e[n1b];
        #pragma unroll
        for (int c = 0; c < 2; ++c) {
            const float bb = c ? bbb : bba;
            const int   hcol = wave * 32 + c * 16 + lcol;
            #pragma unroll
            for (int i = 0; i < 8; ++i) {
                float v0 = hacc[0][c][i] + bb;
                float v1 = hacc[1][c][i] + bb;
                v0 = 0.5f * v0 * (1.0f + erff(v0 * 0.70710678118654752f));
                v1 = 0.5f * v1 * (1.0f + erff(v1 * 0.70710678118654752f));
                hs[mBase + i][hcol]      = f32_to_bf16(v0);
                hs[16 + mBase + i][hcol] = f32_to_bf16(v1);
            }
        }
        __syncthreads();

        // Stream the next weight chunk toward L2 while the matrix pipe works.
        if ((hc + HCHUNK) < HDIM && lane == 0) {
            if (PRECONV) {
                __builtin_prefetch(W1t + (size_t)e * HDIM * DDIM
                                       + (size_t)(hc + HCHUNK + wave * 32) * DDIM, 0, 1);
                __builtin_prefetch(W2t + (size_t)e * DDIM * HDIM
                                       + (size_t)wave * 128 * HDIM + (hc + HCHUNK), 0, 1);
            } else {
                __builtin_prefetch(W1e + (size_t)(hc + HCHUNK), 0, 1);
                __builtin_prefetch(W2e + (size_t)(hc + HCHUNK) * DDIM + wave * HCHUNK, 0, 1);
            }
        }

        // ------- GEMM2: acc += h_chunk @ W2[hc:hc+HCHUNK, :] ----------------
        // flattened (ks, ct) iteration with double-buffered B tile
        {
            const int KSTEPS = HCHUNK / 32;   // 8
            v16bf Bc = load_b_tile<PRECONV>(W2te, W2e, wave * 128 + lcol,
                                            hc + koffB, HDIM, DDIM);
            #pragma unroll
            for (int ks = 0; ks < KSTEPS; ++ks) {
                ABpack A0, A1;
                const int kb = ks * 32 + koffA;
                A0.q[0] = *((const uint4*)&hs[lcol][kb]);
                A0.q[1] = *((const uint4*)&hs[lcol][kb + 8]);
                A1.q[0] = *((const uint4*)&hs[16 + lcol][kb]);
                A1.q[1] = *((const uint4*)&hs[16 + lcol][kb + 8]);
                #pragma unroll
                for (int ct = 0; ct < 8; ++ct) {
                    v16bf Bn = Bc;
                    const int tn = ks * 8 + ct + 1;
                    if (tn < KSTEPS * 8) {
                        const int ksn = tn >> 3, ctn = tn & 7;
                        Bn = load_b_tile<PRECONV>(W2te, W2e,
                                 wave * 128 + ctn * 16 + lcol,
                                 hc + ksn * 32 + koffB, HDIM, DDIM);
                    }
                    acc[0][ct] = __builtin_amdgcn_wmma_f32_16x16x32_bf16(
                                     false, A0.v, false, Bc, (short)0, acc[0][ct], false, false);
                    acc[1][ct] = __builtin_amdgcn_wmma_f32_16x16x32_bf16(
                                     false, A1.v, false, Bc, (short)0, acc[1][ct], false, false);
                    Bc = Bn;
                }
            }
        }
        __syncthreads();   // hs consumed before next chunk overwrites it
    }

    // ------------------------ epilogue: scale + combine ----------------------
    #pragma unroll
    for (int r = 0; r < 2; ++r) {
        #pragma unroll
        for (int ct = 0; ct < 8; ++ct) {
            const int n   = wave * 128 + ct * 16 + lcol;
            const float bb2 = b2e[n];
            #pragma unroll
            for (int i = 0; i < 8; ++i) {
                const int m = r * 16 + mBase + i;
                if (m < rows) {
                    float val = cs[m] * (acc[r][ct][i] + bb2);
                    size_t o  = (size_t)toks[m] * DDIM + n;
                    if (slot == 0) out[o] = xin[o] + val;   // residual + slot0
                    else           out[o] = out[o] + val;   // slot1 (pass B)
                }
            }
        }
    }
}

// ---------------------------------------------------------------------------
extern "C" void kernel_launch(void* const* d_in, const int* in_sizes, int n_in,
                              void* d_out, int out_size, void* d_ws, size_t ws_size,
                              hipStream_t stream) {
    (void)in_sizes; (void)n_in; (void)out_size;
    const float* x   = (const float*)d_in[0];
    const float* ew  = (const float*)d_in[1];
    const float* W1  = (const float*)d_in[2];
    const float* b1  = (const float*)d_in[3];
    const float* W2  = (const float*)d_in[4];
    const float* b2  = (const float*)d_in[5];
    const int*   idx = (const int*)d_in[6];
    float* out = (float*)d_out;

    const size_t xbB    = (size_t)NTOK * DDIM * 2;          // 8 MB
    const size_t metaB  = (size_t)NTOK * 4 * 4 + (size_t)2 * NEXP * NTOK * 4
                        + (size_t)2 * NEXP * 4;             // ~0.3 MB
    const size_t wtB    = (size_t)NEXP * DDIM * HDIM * 2;   // 64 MB each

    char* ws = (char*)d_ws;
    unsigned short* xb = (unsigned short*)ws; ws += xbB;
    int*   e0 = (int*)ws;   ws += (size_t)NTOK * 4;
    int*   e1 = (int*)ws;   ws += (size_t)NTOK * 4;
    float* c0 = (float*)ws; ws += (size_t)NTOK * 4;
    float* c1 = (float*)ws; ws += (size_t)NTOK * 4;
    int* lists  = (int*)ws; ws += (size_t)2 * NEXP * NTOK * 4;
    int* counts = (int*)ws; ws += (size_t)2 * NEXP * 4;

    const bool preconv = ws_size >= (xbB + metaB + 2 * wtB + 256);
    unsigned short* w1t = (unsigned short*)ws; ws += wtB;
    unsigned short* w2t = (unsigned short*)ws; ws += wtB;

    router_kernel<<<NTOK / 256, 256, 0, stream>>>(ew, idx, e0, e1, c0, c1);
    build_lists_kernel<<<1, 16, 0, stream>>>(e0, e1, lists, counts);
    cast_x_kernel<<<256, 256, 0, stream>>>(x, xb);

    if (preconv) {
        // W1 [E][D][H] -> W1t [E][H][D];  W2 [E][H][D] -> W2t [E][D][H]
        transpose_cast_kernel<<<dim3(HDIM / 32, DDIM / 32, NEXP), 256, 0, stream>>>(
            W1, w1t, DDIM, HDIM);
        transpose_cast_kernel<<<dim3(DDIM / 32, HDIM / 32, NEXP), 256, 0, stream>>>(
            W2, w2t, HDIM, DDIM);
    }

    dim3 grid(NEXP * TILES);   // 1024 blocks; empty tile rows exit immediately
    if (preconv) {
        moe_ffn_kernel<true><<<grid, 256, 0, stream>>>(
            0, xb, x, W1, w1t, b1, W2, w2t, b2, lists, counts, c0, c1, out);
        moe_ffn_kernel<true><<<grid, 256, 0, stream>>>(
            1, xb, x, W1, w1t, b1, W2, w2t, b2, lists, counts, c0, c1, out);
    } else {
        moe_ffn_kernel<false><<<grid, 256, 0, stream>>>(
            0, xb, x, W1, w1t, b1, W2, w2t, b2, lists, counts, c0, c1, out);
        moe_ffn_kernel<false><<<grid, 256, 0, stream>>>(
            1, xb, x, W1, w1t, b1, W2, w2t, b2, lists, counts, c0, c1, out);
    }
}
